// LSTMModel_19499151523996
// MI455X (gfx1250) — matrix operand order repaired and checked by reading, not compiled, not run on hardware
//
#include <hip/hip_runtime.h>

typedef __attribute__((ext_vector_type(16))) _Float16 v16h;
typedef __attribute__((ext_vector_type(8)))  _Float16 v8h;
typedef __attribute__((ext_vector_type(8)))  float    v8f;
typedef __attribute__((ext_vector_type(4)))  int      i4;

#define HH   50      // hidden size
#define GG   200     // 4*H gates
#define NP   208     // gate dim padded to 13*16
#define TT   256     // seq len
#define BB   512     // batch
#define II   144     // input features
#define K0P  160     // input K padded to 5*32
#define BT   32      // batch tile per workgroup (phase 2)

#if __has_builtin(__builtin_amdgcn_global_load_async_to_lds_b128) && \
    __has_builtin(__builtin_amdgcn_s_wait_asynccnt)
#define HAVE_ASYNC_LDS 1
#else
#define HAVE_ASYNC_LDS 0
#endif

__device__ __forceinline__ float fsigmoid(float x) {
    return 1.f / (1.f + __expf(-x));
}
__device__ __forceinline__ float ftanh(float x) {
    float xc = fminf(fmaxf(x, -15.f), 15.f);
    float e = __expf(2.f * xc);
    return (e - 1.f) / (e + 1.f);
}

// 16-byte global -> LDS copy: async DMA path on CDNA5, sync fallback otherwise.
__device__ __forceinline__ void copy16_g2l(const _Float16* g, _Float16* l) {
#if HAVE_ASYNC_LDS
    __builtin_amdgcn_global_load_async_to_lds_b128(
        (__attribute__((address_space(1))) i4*)g,
        (__attribute__((address_space(3))) i4*)l,
        0, 0);
#else
    *(v8h*)l = *(const v8h*)g;
#endif
}
__device__ __forceinline__ void wait_async0() {
#if HAVE_ASYNC_LDS
    __builtin_amdgcn_s_wait_asynccnt(0);
#endif
}

// ---------------------------------------------------------------------------
// Phase 1: x_proj = x @ w_ih0^T + (b_ih0 + b_hh0), stored f16 [B*T][200]
// grid = (B*T)/64 blocks, 256 threads (8 waves)
// ---------------------------------------------------------------------------
__global__ void __launch_bounds__(256)
xproj_kernel(const float* __restrict__ x,      // [B*T][144]
             const float* __restrict__ w_ih0,  // [200][144]
             const float* __restrict__ b_ih0,  // [200]
             const float* __restrict__ b_hh0,  // [200]
             _Float16* __restrict__ xproj)     // [B*T][200]
{
    __shared__ __align__(16) _Float16 Ax[64 * K0P];   // 20.5 KB
    __shared__ __align__(16) _Float16 Wt[NP * K0P];   // 66.6 KB
    __shared__ float bias[NP];

    const int  tid = threadIdx.x;
    const long m0  = (long)blockIdx.x * 64;

    for (int i = tid; i < NP * K0P; i += 256) {
        int n = i / K0P, k = i % K0P;
        Wt[i] = (n < GG && k < II) ? (_Float16)w_ih0[n * II + k] : (_Float16)0.f;
    }
    for (int i = tid; i < NP; i += 256)
        bias[i] = (i < GG) ? (b_ih0[i] + b_hh0[i]) : 0.f;
    for (int i = tid; i < 64 * K0P; i += 256) {
        int m = i / K0P, k = i % K0P;
        Ax[i] = (k < II) ? (_Float16)x[(m0 + m) * II + k] : (_Float16)0.f;
    }
    __syncthreads();

    const int wave = tid >> 5, lane = tid & 31;
    const int col  = lane & 15;        // column (B/C/D) / row (A) within tile
    const int hi   = lane >> 4;        // lane-group select (ISA fragment layout)

    for (int tile = wave; tile < 4 * 13; tile += 8) {
        const int mt = tile / 13, nt = tile % 13;
        const float bn = bias[nt * 16 + col];
        v8f c;
        #pragma unroll
        for (int g = 0; g < 8; ++g) c[g] = bn;

        const int arow = mt * 16 + col;
        const int bcol = nt * 16 + col;
        #pragma unroll
        for (int kt = 0; kt < K0P / 32; ++kt) {
            // A 16x32 f16: lanes 0-15 hold K 0..15, lanes 16-31 K 16..31
            const _Float16* ap = &Ax[arow * K0P + kt * 32 + hi * 16];
            v8h a0 = *(const v8h*)(ap);
            v8h a1 = *(const v8h*)(ap + 8);
            v16h a = __builtin_shufflevector(a0, a1, 0,1,2,3,4,5,6,7,8,9,10,11,12,13,14,15);
            // B 32x16 f16: lane holds col, 16 contiguous K (lo:0-15, hi:16-31)
            v16h b = *(const v16h*)(&Wt[bcol * K0P + kt * 32 + hi * 16]);
            c = __builtin_amdgcn_wmma_f32_16x16x32_f16(false, a, false, b,
                                                       (short)0, c, false, false);
        }
        if (bcol < GG) {
            #pragma unroll
            for (int g = 0; g < 8; ++g) {
                long row = m0 + mt * 16 + g + hi * 8;   // C/D: row = g + 8*hi
                xproj[row * GG + bcol] = (_Float16)c[g];
            }
        }
    }
}

// ---------------------------------------------------------------------------
// Phase 2: fused 2-layer LSTM recurrence + final FC.
// grid = B/BT = 16 blocks, 256 threads (8 waves). All state lives in LDS.
//  xg   f16 [2][BT][200] : double-buffered x_proj tile (async global->LDS)
//  A2   f16 [BT][128]    : cols 0..49 = h1, 64..113 = h2 (pads stay zero)
//  W2   f16 [NP][128]    : cols 0..49 = w_ih1, 64..113 = w_hh1 (fused GEMM)
// ---------------------------------------------------------------------------
__global__ void __launch_bounds__(256)
lstm_rec_kernel(const _Float16* __restrict__ xproj, // [B*T][200]
                const float* __restrict__ w_hh0,    // [200][50]
                const float* __restrict__ w_ih1,    // [200][50]
                const float* __restrict__ w_hh1,    // [200][50]
                const float* __restrict__ b_ih1,    // [200]
                const float* __restrict__ b_hh1,    // [200]
                const float* __restrict__ w_fc,     // [144][50]
                const float* __restrict__ b_fc,     // [144]
                float* __restrict__ out)            // [512][144]
{
    __shared__ __align__(16) _Float16 xg[2][BT * GG]; // 25.6 KB (12.8 x2)
    __shared__ float    gates[BT * NP];               // 26.6 KB
    __shared__ __align__(16) _Float16 A2[BT * 128];   //  8.0 KB
    __shared__ __align__(16) _Float16 W0[NP * 64];    // 26.6 KB
    __shared__ __align__(16) _Float16 W2[NP * 128];   // 53.2 KB
    __shared__ float    bias2[NP];
    __shared__ float    c1[BT * HH];
    __shared__ float    c2[BT * HH];

    const int tid = threadIdx.x;
    const int b0  = blockIdx.x * BT;

    for (int i = tid; i < NP * 64; i += 256) {
        int n = i >> 6, k = i & 63;
        W0[i] = (n < GG && k < HH) ? (_Float16)w_hh0[n * HH + k] : (_Float16)0.f;
    }
    for (int i = tid; i < NP * 128; i += 256) {
        int n = i >> 7, k = i & 127;
        float v = 0.f;
        if (n < GG) {
            if (k < HH)                      v = w_ih1[n * HH + k];
            else if (k >= 64 && k < 64 + HH) v = w_hh1[n * HH + (k - 64)];
        }
        W2[i] = (_Float16)v;
    }
    for (int i = tid; i < NP; i += 256)
        bias2[i] = (i < GG) ? (b_ih1[i] + b_hh1[i]) : 0.f;
    for (int i = tid; i < BT * 128; i += 256) A2[i] = (_Float16)0.f;
    for (int i = tid; i < BT * HH; i += 256) { c1[i] = 0.f; c2[i] = 0.f; }

    // Prime the pipeline: async-copy tile t=0 into buffer 0.
    // Tile = 32 rows x 400 B; 25 16-byte chunks per row, 800 chunks total.
    for (int i = tid; i < BT * 25; i += 256) {
        int b = i / 25, o = i % 25;
        copy16_g2l(&xproj[((long)(b0 + b) * TT + 0) * GG + o * 8],
                   &xg[0][b * GG + o * 8]);
    }
    __syncthreads();

    const int wave = tid >> 5, lane = tid & 31;
    const int col  = lane & 15, hi = lane >> 4;

    for (int t = 0; t < TT; ++t) {
        const int cur = t & 1;

        // buffer `cur` (issued one step ago) is complete; prev-step LDS settled
        wait_async0();
        __syncthreads();

        // kick off next step's tile copy (overlaps both GEMMs + activations)
        if (t + 1 < TT) {
            for (int i = tid; i < BT * 25; i += 256) {
                int b = i / 25, o = i % 25;
                copy16_g2l(&xproj[((long)(b0 + b) * TT + (t + 1)) * GG + o * 8],
                           &xg[cur ^ 1][b * GG + o * 8]);
            }
        }

        // --- layer-0 GEMM: gates = xproj_tile + h1 @ w_hh0^T ----------------
        for (int tile = wave; tile < 2 * 13; tile += 8) {
            const int mt = tile / 13, nt = tile % 13;
            const int cn = nt * 16 + col;
            v8f c;
            #pragma unroll
            for (int g = 0; g < 8; ++g)
                c[g] = (cn < GG) ? (float)xg[cur][(mt * 16 + g + hi * 8) * GG + cn]
                                 : 0.f;
            #pragma unroll
            for (int kt = 0; kt < 2; ++kt) {
                const _Float16* ap = &A2[(mt * 16 + col) * 128 + kt * 32 + hi * 16];
                v8h a0 = *(const v8h*)(ap);
                v8h a1 = *(const v8h*)(ap + 8);
                v16h a = __builtin_shufflevector(a0, a1, 0,1,2,3,4,5,6,7,8,9,10,11,12,13,14,15);
                v16h b = *(const v16h*)(&W0[cn * 64 + kt * 32 + hi * 16]);
                c = __builtin_amdgcn_wmma_f32_16x16x32_f16(false, a, false, b,
                                                           (short)0, c, false, false);
            }
            #pragma unroll
            for (int g = 0; g < 8; ++g)
                gates[(mt * 16 + g + hi * 8) * NP + cn] = c[g];
        }
        __syncthreads();

        // --- layer-0 pointwise: update c1, h1 -> A2[:,0:50] -----------------
        for (int idx = tid; idx < BT * HH; idx += 256) {
            int b = idx / HH, j = idx % HH;
            float ig = fsigmoid(gates[b * NP + j]);
            float fg = fsigmoid(gates[b * NP + HH + j]);
            float gg = ftanh   (gates[b * NP + 2 * HH + j]);
            float og = fsigmoid(gates[b * NP + 3 * HH + j]);
            float cc = fg * c1[idx] + ig * gg;
            c1[idx] = cc;
            A2[b * 128 + j] = (_Float16)(og * ftanh(cc));
        }
        __syncthreads();

        // --- layer-1 fused GEMM: gates = bias1 + [h1|h2] @ [w_ih1|w_hh1]^T --
        for (int tile = wave; tile < 2 * 13; tile += 8) {
            const int mt = tile / 13, nt = tile % 13;
            const int cn = nt * 16 + col;
            const float bn = bias2[cn];
            v8f c;
            #pragma unroll
            for (int g = 0; g < 8; ++g) c[g] = bn;
            #pragma unroll
            for (int kt = 0; kt < 4; ++kt) {
                const _Float16* ap = &A2[(mt * 16 + col) * 128 + kt * 32 + hi * 16];
                v8h a0 = *(const v8h*)(ap);
                v8h a1 = *(const v8h*)(ap + 8);
                v16h a = __builtin_shufflevector(a0, a1, 0,1,2,3,4,5,6,7,8,9,10,11,12,13,14,15);
                v16h b = *(const v16h*)(&W2[cn * 128 + kt * 32 + hi * 16]);
                c = __builtin_amdgcn_wmma_f32_16x16x32_f16(false, a, false, b,
                                                           (short)0, c, false, false);
            }
            #pragma unroll
            for (int g = 0; g < 8; ++g)
                gates[(mt * 16 + g + hi * 8) * NP + cn] = c[g];
        }
        __syncthreads();

        // --- layer-1 pointwise: update c2, h2 -> A2[:,64:114] ---------------
        for (int idx = tid; idx < BT * HH; idx += 256) {
            int b = idx / HH, j = idx % HH;
            float ig = fsigmoid(gates[b * NP + j]);
            float fg = fsigmoid(gates[b * NP + HH + j]);
            float gg = ftanh   (gates[b * NP + 2 * HH + j]);
            float og = fsigmoid(gates[b * NP + 3 * HH + j]);
            float cc = fg * c2[idx] + ig * gg;
            c2[idx] = cc;
            A2[b * 128 + 64 + j] = (_Float16)(og * ftanh(cc));
        }
        __syncthreads();
    }

    // --- final FC: out = h2_last @ w_fc^T + b_fc (tiny, scalar from LDS) ----
    for (int idx = tid; idx < BT * II; idx += 256) {
        int b = idx / II, o = idx % II;
        float s = b_fc[o];
        #pragma unroll 5
        for (int k = 0; k < HH; ++k)
            s += (float)A2[b * 128 + 64 + k] * w_fc[o * HH + k];
        out[(long)(b0 + b) * II + o] = s;
    }
}

// ---------------------------------------------------------------------------
extern "C" void kernel_launch(void* const* d_in, const int* in_sizes, int n_in,
                              void* d_out, int out_size, void* d_ws, size_t ws_size,
                              hipStream_t stream) {
    (void)in_sizes; (void)n_in; (void)out_size; (void)ws_size;

    const float* x     = (const float*)d_in[0];
    const float* w_ih0 = (const float*)d_in[1];
    const float* w_hh0 = (const float*)d_in[2];
    const float* b_ih0 = (const float*)d_in[3];
    const float* b_hh0 = (const float*)d_in[4];
    const float* w_ih1 = (const float*)d_in[5];
    const float* w_hh1 = (const float*)d_in[6];
    const float* b_ih1 = (const float*)d_in[7];
    const float* b_hh1 = (const float*)d_in[8];
    const float* w_fc  = (const float*)d_in[9];
    const float* b_fc  = (const float*)d_in[10];
    float*       out   = (float*)d_out;

    _Float16* xproj = (_Float16*)d_ws;   // [B*T][200] f16 = 52.4 MB

    // Phase 1: big parallel input projection (WMMA f16 -> f32 acc)
    xproj_kernel<<<(BB * TT) / 64, 256, 0, stream>>>(x, w_ih0, b_ih0, b_hh0, xproj);

    // Phase 2: serial recurrence, batch-parallel across 16 workgroups,
    // next-step x_proj tile streamed in via async global->LDS DMA.
    lstm_rec_kernel<<<BB / BT, 256, 0, stream>>>(xproj, w_hh0, w_ih1, w_hh1,
                                                 b_ih1, b_hh1, w_fc, b_fc, out);
}
